// Model_39676907884013
// MI455X (gfx1250) — matrix operand order repaired and checked by reading, not compiled, not run on hardware
//
#include <hip/hip_runtime.h>

// ---------------------------------------------------------------------------
// Fused MHA block for MI455X (gfx1250, wave32, WMMA bf16 16x16x32, f32 accum)
// B=2, S=2048, E=1024, H=16, hd=64. Reference multiplies scores by sqrt(hd)=8.
// Workspace layout (48 MB total):
//   [ 0MB) qbf     : query as bf16        (B*S, E)          8 MB
//   [ 8MB) wqkvT   : Wqkv^T as bf16       (3E, E)           6 MB
//   [14MB) wprojT  : Wproj^T as bf16      (E, E)            2 MB
//   [16MB) Qb      : Q bf16               (B,H,S,hd)        8 MB
//   [24MB) Kb      : K bf16               (B,H,S,hd)        8 MB
//   [32MB) Vtb     : V^T bf16             (B,H,hd,S)        8 MB
//   [40MB) attnbf  : attn output bf16     (B*S, E)          8 MB
// ---------------------------------------------------------------------------

typedef __attribute__((ext_vector_type(16))) __bf16       bf16x16;
typedef __attribute__((ext_vector_type(8)))  float        f32x8;
typedef __attribute__((ext_vector_type(8)))  unsigned int u32x8;

union FragAB { u32x8 u; bf16x16 bf; };

__device__ __forceinline__ unsigned short f2bf(float f) {
  union { float f; unsigned u; } v; v.f = f;
  unsigned r = v.u + 0x7FFFu + ((v.u >> 16) & 1u);   // round-to-nearest-even
  return (unsigned short)(r >> 16);
}

__device__ __forceinline__ f32x8 zero8() {
  f32x8 z;
#pragma unroll
  for (int i = 0; i < 8; ++i) z[i] = 0.f;
  return z;
}

// Compiler-only memory barrier: HW keeps same-wave LDS ops in order (ISA §7.3),
// we only need to stop clang reordering differently-typed LDS accesses.
__device__ __forceinline__ void compiler_fence() {
  asm volatile("" ::: "memory");
}

// A-fragment (16x32 bf16, row-major source, 2 bf16/dword).
// lane half g: VGPR v<4 -> K=2v+8g ; v>=4 -> K=16+2(v-4)+8g
// Byte offsets on a uniform char* base => addr = s64 + zext(v32), so the
// backend selects global_load_b128 vdst, voff, s[base] offset:imm (GVS mode).
__device__ __forceinline__ FragAB load_a(const char* base,
                                         unsigned byteoff, unsigned g) {
  uint4 lo = *reinterpret_cast<const uint4*>(base + (byteoff + g * 16u));
  uint4 hi = *reinterpret_cast<const uint4*>(base + (byteoff + 32u + g * 16u));
  FragAB a;
  a.u[0] = lo.x; a.u[1] = lo.y; a.u[2] = lo.z; a.u[3] = lo.w;
  a.u[4] = hi.x; a.u[5] = hi.y; a.u[6] = hi.z; a.u[7] = hi.w;
  return a;
}

// B-fragment (32x16 bf16). Column n = lane&15 ; lane half g: v<4 -> K=2v+8g ; v>=4 -> K=16+2(v-4)+8g
// Source has K contiguous per output-column row ("transposed" weights).
__device__ __forceinline__ FragAB load_b(const char* base,
                                         unsigned byteoff, unsigned g) {
  uint4 lo = *reinterpret_cast<const uint4*>(base + (byteoff + g * 16u));
  uint4 hi = *reinterpret_cast<const uint4*>(base + (byteoff + 32u + g * 16u));
  FragAB b;
  b.u[0] = lo.x; b.u[1] = lo.y; b.u[2] = lo.z; b.u[3] = lo.w;
  b.u[4] = hi.x; b.u[5] = hi.y; b.u[6] = hi.z; b.u[7] = hi.w;
  return b;
}

__device__ __forceinline__ f32x8 wmma_bf16(FragAB a, FragAB b, f32x8 c) {
  return __builtin_amdgcn_wmma_f32_16x16x32_bf16(
      false, a.bf, false, b.bf, (short)0, c, false, false);
}

// ------------------------------- converters --------------------------------
__global__ void k_cvt_bf16(const float* __restrict__ in,
                           unsigned short* __restrict__ out, int n) {
  int i = blockIdx.x * 256 + threadIdx.x;
  if (i < n) out[i] = f2bf(in[i]);
}

// in: (rows x cols) fp32 row-major ; out: (cols x rows) bf16 row-major
__global__ void k_transpose_bf16(const float* __restrict__ in,
                                 unsigned short* __restrict__ out,
                                 int rows, int cols) {
  int o = blockIdx.x * 256 + threadIdx.x;
  if (o >= rows * cols) return;
  int j = o / rows;          // output row = input column
  int i = o % rows;          // output col = input row
  out[o] = f2bf(in[(size_t)i * cols + j]);
}

// ------------------------------- QKV GEMM ----------------------------------
// C(4096 x 3072) = qbf(4096 x 1024) @ Wqkv + bqkv, scattered to Q/K/V^T bf16.
// Wave tile 32x64. Manually 2x-unrolled ping-pong buffers: loads for kc+1 fly
// while WMMAs for kc execute, with NO inter-buffer register copies.
__global__ __launch_bounds__(128) void
k_qkv(const char* __restrict__ qc,      // bf16 (4096 x 2048 B rows)
      const char* __restrict__ wc,      // bf16 (3072 x 2048 B rows)
      const float* __restrict__ bias,
      unsigned short* __restrict__ Qb,
      unsigned short* __restrict__ Kb,
      unsigned short* __restrict__ Vt) {
  const unsigned lane = threadIdx.x & 31u, w = threadIdx.x >> 5;
  const unsigned g = lane >> 4, ln = lane & 15u;
  const unsigned m0 = blockIdx.x * 128u + w * 32u;
  const unsigned n0 = blockIdx.y * 64u;

  f32x8 acc[2][4];
#pragma unroll
  for (int rb = 0; rb < 2; ++rb)
#pragma unroll
    for (int nt = 0; nt < 4; ++nt) acc[rb][nt] = zero8();

  const unsigned aoff0 = (m0 + ln) * 2048u;
  const unsigned aoff1 = (m0 + 16u + ln) * 2048u;
  unsigned boff[4];
#pragma unroll
  for (int nt = 0; nt < 4; ++nt) boff[nt] = (n0 + nt * 16u + ln) * 2048u;

  FragAB A0[2], A1[2], Bf[2][4];
  A0[0] = load_a(qc, aoff0, g);
  A1[0] = load_a(qc, aoff1, g);
#pragma unroll
  for (int nt = 0; nt < 4; ++nt) Bf[0][nt] = load_b(wc, boff[nt], g);

  auto step = [&](unsigned kc, int cur, int nxt) {
    const unsigned kn = ((kc + 1u) & 31u) * 64u;   // 16 dw = 64 B per K chunk
#pragma unroll
    for (int nt = 0; nt < 4; ++nt)
      __builtin_prefetch(wc + boff[nt] + ((kc + 2u) & 31u) * 64u, 0, 3);
    A0[nxt] = load_a(qc, aoff0 + kn, g);
    A1[nxt] = load_a(qc, aoff1 + kn, g);
#pragma unroll
    for (int nt = 0; nt < 4; ++nt) Bf[nxt][nt] = load_b(wc, boff[nt] + kn, g);
#pragma unroll
    for (int nt = 0; nt < 4; ++nt) {
      acc[0][nt] = wmma_bf16(A0[cur], Bf[cur][nt], acc[0][nt]);
      acc[1][nt] = wmma_bf16(A1[cur], Bf[cur][nt], acc[1][nt]);
    }
  };

  for (unsigned kc = 0; kc < 32; kc += 2) {
    step(kc, 0, 1);
    step(kc + 1u, 1, 0);
  }

#pragma unroll
  for (int nt = 0; nt < 4; ++nt) {
    const unsigned col = n0 + nt * 16u + ln;   // global column in [0, 3072)
    const float bv = bias[col];
    const unsigned h = col / 192u, c = col % 192u;  // head, offset (q|k|v)
#pragma unroll
    for (int rb = 0; rb < 2; ++rb) {
#pragma unroll
      for (int r = 0; r < 8; ++r) {
        const unsigned m = m0 + rb * 16u + r + 8u * g;  // global row (b*S+s)
        const unsigned bb = m >> 11, s = m & 2047u;
        const unsigned short o = f2bf(acc[rb][nt][r] + bv);
        if (c < 64u)
          Qb[((bb * 16u + h) * 2048u + s) * 64u + c] = o;
        else if (c < 128u)
          Kb[((bb * 16u + h) * 2048u + s) * 64u + (c - 64u)] = o;
        else
          Vt[((bb * 16u + h) * 64u + (c - 128u)) * 2048u + s] = o;
      }
    }
  }
}

// --------------------------- flash attention -------------------------------
// Block = 4 waves, each wave owns 16 query rows; grid (B*H, S/64).
// Pipelined + 2x-unrolled: K frags for chunk c+1 load (into the other buffer)
// during softmax of chunk c; V frags for chunk c are issued before softmax so
// exp/shfl VALU covers their latency. No buffer-rotation copies.
__global__ __launch_bounds__(128) void
k_attn(const char* __restrict__ Qc,     // (BH*2048) rows x 128 B
       const char* __restrict__ Kc,
       const char* __restrict__ Vc,     // (BH*64) rows x 4096 B
       unsigned short* __restrict__ attn) {   // (4096 x 1024) bf16
  __shared__ unsigned int Plds[4][16 * 16];   // per-wave 16x32 bf16 P tile

  const unsigned lane = threadIdx.x & 31u, w = threadIdx.x >> 5;
  const unsigned g = lane >> 4, ln = lane & 15u;
  const unsigned bh = blockIdx.x, b = bh >> 4, h = bh & 15u;
  const unsigned q0 = blockIdx.y * 64u + w * 16u;
  const unsigned rbase = bh * 2048u;

  const unsigned qoff = (rbase + q0 + ln) * 128u;
  const FragAB aq0 = load_a(Qc, qoff, g);
  const FragAB aq1 = load_a(Qc, qoff + 64u, g);

  const unsigned koff0 = (rbase + ln) * 128u;          // + key*128 B
  const unsigned koff1 = (rbase + 16u + ln) * 128u;
  unsigned voff[4];
#pragma unroll
  for (int nt = 0; nt < 4; ++nt)
    voff[nt] = (bh * 64u + nt * 16u + ln) * 4096u;

  f32x8 o[4];
#pragma unroll
  for (int nt = 0; nt < 4; ++nt) o[nt] = zero8();
  float mr[8], lr[8];
#pragma unroll
  for (int r = 0; r < 8; ++r) { mr[r] = -3.0e38f; lr[r] = 0.f; }

  unsigned short* Pw = (unsigned short*)&Plds[w][0];

  // K fragment ping-pong buffers: kf[buf][t][c] (key sub-tile t, K-half c)
  FragAB kf[2][2][2];
#pragma unroll
  for (int t = 0; t < 2; ++t) {
    const unsigned ko = (t ? koff1 : koff0);
    kf[0][t][0] = load_b(Kc, ko, g);
    kf[0][t][1] = load_b(Kc, ko + 64u, g);
  }

  auto step = [&](unsigned ch, int cur, int nxt) {
    // ---- issue V fragment loads for this chunk (consumed after softmax) ----
    FragAB vf[4];
#pragma unroll
    for (int nt = 0; nt < 4; ++nt) vf[nt] = load_b(Vc, voff[nt] + ch * 64u, g);
    // ---- issue K fragment loads for the next chunk ----
    const unsigned chn = (ch + 1u) & 63u;
#pragma unroll
    for (int t = 0; t < 2; ++t) {
      const unsigned ko = (t ? koff1 : koff0) + chn * 4096u;  // 32 keys*128 B
      kf[nxt][t][0] = load_b(Kc, ko, g);
      kf[nxt][t][1] = load_b(Kc, ko + 64u, g);
    }
    // prefetch V rows one chunk ahead
#pragma unroll
    for (int nt = 0; nt < 4; ++nt)
      __builtin_prefetch(Vc + voff[nt] + chn * 64u, 0, 3);

    // ---- scores S = 8 * Q K^T for 32 keys (K frags already resident) ----
    f32x8 sc[2];
#pragma unroll
    for (int t = 0; t < 2; ++t) {
      f32x8 s = zero8();
      s = wmma_bf16(aq0, kf[cur][t][0], s);
      s = wmma_bf16(aq1, kf[cur][t][1], s);
      sc[t] = s;
    }

    // ---- online softmax over 16-lane row groups (covers V/K latency) ----
#pragma unroll
    for (int r = 0; r < 8; ++r) {
      float s0 = sc[0][r] * 8.0f;
      float s1 = sc[1][r] * 8.0f;
      float mx = fmaxf(s0, s1);
#pragma unroll
      for (int off = 1; off < 16; off <<= 1)
        mx = fmaxf(mx, __shfl_xor(mx, off, 16));
      const float mnew = fmaxf(mr[r], mx);
      const float p0 = __expf(s0 - mnew);
      const float p1 = __expf(s1 - mnew);
      float rs = p0 + p1;
#pragma unroll
      for (int off = 1; off < 16; off <<= 1)
        rs += __shfl_xor(rs, off, 16);
      const float alpha = __expf(mr[r] - mnew);
      lr[r] = lr[r] * alpha + rs;
      mr[r] = mnew;
#pragma unroll
      for (int nt = 0; nt < 4; ++nt) o[nt][r] *= alpha;
      const unsigned m = r + 8u * g;             // C-layout row
      Pw[m * 32u + ln]       = f2bf(p0);
      Pw[m * 32u + 16u + ln] = f2bf(p1);
    }
    compiler_fence();  // P tile is wave-private; same-wave DS stays in order
    // ---- P (16x32) as A-fragment from LDS ----
    const FragAB ap = load_a((const char*)&Plds[w][0], ln * 64u, g);
    compiler_fence();  // keep next iteration's P stores after these gathers
    // ---- O += P @ V (4 column tiles of 16) ----
#pragma unroll
    for (int nt = 0; nt < 4; ++nt) o[nt] = wmma_bf16(ap, vf[nt], o[nt]);
  };

  for (unsigned ch = 0; ch < 64; ch += 2) {
    step(ch, 0, 1);
    step(ch + 1u, 1, 0);
  }

  // ---- epilogue: O /= l, write bf16 (B,S,E) ----
#pragma unroll
  for (int r = 0; r < 8; ++r) {
    const float inv = 1.0f / lr[r];
    const unsigned srow = q0 + r + 8u * g;
#pragma unroll
    for (int nt = 0; nt < 4; ++nt) {
      const unsigned col = h * 64u + nt * 16u + ln;
      attn[(b * 2048u + srow) * 1024u + col] = f2bf(o[nt][r] * inv);
    }
  }
}

// ------------------------------ projection ---------------------------------
// Same 32x64 ping-pong scheme as k_qkv; fp32 output + bias.
__global__ __launch_bounds__(128) void
k_proj(const char* __restrict__ ac,     // bf16 (4096 x 2048 B rows)
       const char* __restrict__ wc,     // bf16 (1024 x 2048 B rows)
       const float* __restrict__ bias,
       float* __restrict__ out) {       // (4096 x 1024) fp32
  const unsigned lane = threadIdx.x & 31u, w = threadIdx.x >> 5;
  const unsigned g = lane >> 4, ln = lane & 15u;
  const unsigned m0 = blockIdx.x * 128u + w * 32u;
  const unsigned n0 = blockIdx.y * 64u;

  f32x8 acc[2][4];
#pragma unroll
  for (int rb = 0; rb < 2; ++rb)
#pragma unroll
    for (int nt = 0; nt < 4; ++nt) acc[rb][nt] = zero8();

  const unsigned aoff0 = (m0 + ln) * 2048u;
  const unsigned aoff1 = (m0 + 16u + ln) * 2048u;
  unsigned boff[4];
#pragma unroll
  for (int nt = 0; nt < 4; ++nt) boff[nt] = (n0 + nt * 16u + ln) * 2048u;

  FragAB A0[2], A1[2], Bf[2][4];
  A0[0] = load_a(ac, aoff0, g);
  A1[0] = load_a(ac, aoff1, g);
#pragma unroll
  for (int nt = 0; nt < 4; ++nt) Bf[0][nt] = load_b(wc, boff[nt], g);

  auto step = [&](unsigned kc, int cur, int nxt) {
    const unsigned kn = ((kc + 1u) & 31u) * 64u;
#pragma unroll
    for (int nt = 0; nt < 4; ++nt)
      __builtin_prefetch(wc + boff[nt] + ((kc + 2u) & 31u) * 64u, 0, 3);
    A0[nxt] = load_a(ac, aoff0 + kn, g);
    A1[nxt] = load_a(ac, aoff1 + kn, g);
#pragma unroll
    for (int nt = 0; nt < 4; ++nt) Bf[nxt][nt] = load_b(wc, boff[nt] + kn, g);
#pragma unroll
    for (int nt = 0; nt < 4; ++nt) {
      acc[0][nt] = wmma_bf16(A0[cur], Bf[cur][nt], acc[0][nt]);
      acc[1][nt] = wmma_bf16(A1[cur], Bf[cur][nt], acc[1][nt]);
    }
  };

  for (unsigned kc = 0; kc < 32; kc += 2) {
    step(kc, 0, 1);
    step(kc + 1u, 1, 0);
  }

#pragma unroll
  for (int nt = 0; nt < 4; ++nt) {
    const unsigned col = n0 + nt * 16u + ln;
    const float bv = bias[col];
#pragma unroll
    for (int rb = 0; rb < 2; ++rb) {
#pragma unroll
      for (int r = 0; r < 8; ++r) {
        const unsigned m = m0 + rb * 16u + r + 8u * g;
        out[(size_t)m * 1024u + col] = acc[rb][nt][r] + bv;
      }
    }
  }
}

// ------------------------------- launcher ----------------------------------
extern "C" void kernel_launch(void* const* d_in, const int* in_sizes, int n_in,
                              void* d_out, int out_size, void* d_ws,
                              size_t ws_size, hipStream_t stream) {
  // setup_inputs order: query, key, value, Wqkv, bqkv, Wproj, bproj
  const float* query = (const float*)d_in[0];
  // d_in[1] (key) and d_in[2] (value) are unused by the reference.
  const float* Wqkv  = (const float*)d_in[3];
  const float* bqkv  = (const float*)d_in[4];
  const float* Wproj = (const float*)d_in[5];
  const float* bproj = (const float*)d_in[6];

  const size_t MB = 1024 * 1024;
  char* ws = (char*)d_ws;
  unsigned short* qbf    = (unsigned short*)(ws + 0 * MB);
  unsigned short* wqkvT  = (unsigned short*)(ws + 8 * MB);
  unsigned short* wprojT = (unsigned short*)(ws + 14 * MB);
  unsigned short* Qb     = (unsigned short*)(ws + 16 * MB);
  unsigned short* Kb     = (unsigned short*)(ws + 24 * MB);
  unsigned short* Vtb    = (unsigned short*)(ws + 32 * MB);
  unsigned short* attnbf = (unsigned short*)(ws + 40 * MB);

  // 1) fp32 -> bf16 conversions / transposes
  k_cvt_bf16<<<4194304 / 256, 256, 0, stream>>>(query, qbf, 4194304);
  k_transpose_bf16<<<(3145728 + 255) / 256, 256, 0, stream>>>(Wqkv, wqkvT,
                                                              1024, 3072);
  k_transpose_bf16<<<(1048576 + 255) / 256, 256, 0, stream>>>(Wproj, wprojT,
                                                              1024, 1024);
  // 2) QKV projection + scatter to Q / K / V^T
  k_qkv<<<dim3(32, 48), 128, 0, stream>>>(
      (const char*)qbf, (const char*)wqkvT, bqkv, Qb, Kb, Vtb);
  // 3) flash attention
  k_attn<<<dim3(32, 32), 128, 0, stream>>>(
      (const char*)Qb, (const char*)Kb, (const char*)Vtb, attnbf);
  // 4) output projection
  k_proj<<<dim3(32, 16), 128, 0, stream>>>(
      (const char*)attnbf, (const char*)wprojT, bproj, (float*)d_out);
}
